// FBRNN_38740605010383
// MI455X (gfx1250) — matrix operand order repaired and checked by reading, not compiled, not run on hardware
//
#include <hip/hip_runtime.h>
#include <stdint.h>

// Problem dims (from reference)
#define TSTEPS 512
#define BATCH  64
#define EMB    512
#define HID    512
#define NLAYER 4
#define ADIM   256
#define G3     1536            // 3*HID
#define BH     (BATCH*HID)     // 32768 = 2^15

// Persistent grid: 16 blocks x 256 threads (wave32 -> 128 waves)
#define NBLK     16
#define BLOCK    256
#define NTHREADS (NBLK*BLOCK)  // 4096
#define NWAVES   (NTHREADS/32) // 128

#define LPITCH   520           // LDS row pitch (bf16 elems); rows = 1040B = 65*16B

typedef __attribute__((ext_vector_type(16))) __bf16 v16bf;
typedef __attribute__((ext_vector_type(8)))  float  v8f;
typedef __attribute__((ext_vector_type(4)))  int    v4i;

// Address-space qualified pointers for the async-to-LDS builtin:
// signature is (v4i addrspace(1)* src, v4i addrspace(3)* dst, imm offset, imm cpol)
typedef __attribute__((address_space(1))) v4i* g_v4i_p;
typedef __attribute__((address_space(3))) v4i* l_v4i_p;

#if defined(__has_builtin)
#if __has_builtin(__builtin_amdgcn_global_load_async_to_lds_b128)
#define USE_ASYNC_LDS 1
#endif
#endif

union FragU { v16bf v; unsigned u[8]; };

__device__ __forceinline__ unsigned short f2bf(float f) {
  unsigned x = __float_as_uint(f);
  unsigned r = x + 0x7fffu + ((x >> 16) & 1u);   // round-to-nearest-even
  return (unsigned short)(r >> 16);
}
__device__ __forceinline__ unsigned pack2bf(float lo, float hi) {
  return (unsigned)f2bf(lo) | ((unsigned)f2bf(hi) << 16);
}
__device__ __forceinline__ float sigm(float x) { return 1.0f / (1.0f + __expf(-x)); }

// Global generic pointer -> addrspace(1) v4i pointer.
__device__ __forceinline__ g_v4i_p to_g128(const void* p) {
  return (g_v4i_p)(uintptr_t)p;
}
// Generic LDS pointer -> wave-relative LDS address (ISA 10.2: low 32 bits).
__device__ __forceinline__ l_v4i_p to_l128(const void* p) {
  return (l_v4i_p)(unsigned)(uintptr_t)p;
}
__device__ __forceinline__ void wait_async0() {
#if defined(__has_builtin) && __has_builtin(__builtin_amdgcn_s_wait_asynccnt)
  __builtin_amdgcn_s_wait_asynccnt(0);
#else
  asm volatile("s_wait_asynccnt 0x0" ::: "memory");
#endif
}

// A-fragment 16x32 bf16 (MxK) from an LDS tile (row pitch LPITCH):
// lane = hi*16+m; VGPR v holds K-pair k = (v/4)*16 + hi*8 + (v%4)*2.
__device__ __forceinline__ v16bf load_fragA_lds(const unsigned short* tile, int k0, int lane) {
  const int m = lane & 15, hi = lane >> 4;
  const unsigned short* p = tile + m * LPITCH + k0;
  FragU f;
#pragma unroll
  for (int v = 0; v < 8; ++v) {
    int k = ((v >> 2) << 4) + hi * 8 + ((v & 3) << 1);
    f.u[v] = *reinterpret_cast<const unsigned*>(p + k);
  }
  return f.v;
}

// A-fragment from global row-major (lda elems) - used by attention GEMMs.
__device__ __forceinline__ v16bf load_fragA_g(const unsigned short* tileA, int lda,
                                              int k0, int lane) {
  const int m = lane & 15, hi = lane >> 4;
  const unsigned short* p = tileA + m * lda + k0;
  FragU f;
#pragma unroll
  for (int v = 0; v < 8; ++v) {
    int k = ((v >> 2) << 4) + hi * 8 + ((v & 3) << 1);
    f.u[v] = *reinterpret_cast<const unsigned*>(p + k);
  }
  return f.v;
}

// B-fragment from pre-swizzled weights: entry (slice,lane) is 8 contiguous u32
// holding K-pairs k = kb*32 + (v/4)*16 + hi*8 + (v%4)*2 for column n = lane&15 (fully coalesced).
__device__ __forceinline__ v16bf load_fragW(const unsigned* base, int slice, int lane) {
  const unsigned* p = base + ((size_t)slice * 32 + lane) * 8;
  FragU f;
#pragma unroll
  for (int v = 0; v < 8; ++v) f.u[v] = p[v];
  return f.v;
}

__device__ __forceinline__ v8f wmma_bf16(v16bf a, v16bf b, v8f c) {
  return __builtin_amdgcn_wmma_f32_16x16x32_bf16(false, a, false, b,
                                                 (short)0, c, false, false);
}

// Device-wide barrier: monotonic counter, all NBLK blocks resident.
__device__ __forceinline__ void grid_sync(unsigned* cnt, unsigned* gen) {
  __threadfence();
  __syncthreads();
  if (threadIdx.x == 0) {
    unsigned g = ++(*gen);
    unsigned target = g * (unsigned)NBLK;
    __hip_atomic_fetch_add(cnt, 1u, __ATOMIC_RELEASE, __HIP_MEMORY_SCOPE_AGENT);
    while (__hip_atomic_load(cnt, __ATOMIC_ACQUIRE, __HIP_MEMORY_SCOPE_AGENT) < target)
      __builtin_amdgcn_s_sleep(2);
  }
  __syncthreads();
  __threadfence();
}

// ---------------- prep: swizzle weights into fragment order, zero state -----------
// wsw entry index: (((l*2+g)*96 + nt)*16 + kb)*32 + lane  -> 8 u32 (32B)
// wasw entry index: ((l*16 + nt)*16 + kb)*32 + lane       -> 8 u32
__global__ void fbrnn_prep(const float* __restrict__ W_ih, const float* __restrict__ W_hh,
                           const float* __restrict__ Wa,
                           unsigned* __restrict__ wsw, unsigned* __restrict__ wasw,
                           float* __restrict__ hf, unsigned short* __restrict__ hbf,
                           unsigned* __restrict__ cnt) {
  const int gid = blockIdx.x * blockDim.x + threadIdx.x;
  const int stride = gridDim.x * blockDim.x;
  if (gid == 0) *cnt = 0u;

  // GRU weights: 4 layers x {ih,hh} x 96 ntile x 16 kb x 32 lanes
  const int NSWE = NLAYER * 2 * 96 * 16 * 32;          // 393,216 entries
  for (int e = gid; e < NSWE; e += stride) {
    int lane = e & 31;
    int kb = (e >> 5) & 15;
    int q = e >> 9;
    int nt = q % 96;
    int lg = q / 96;
    int g = lg & 1, l = lg >> 1;
    int n = lane & 15, hi = lane >> 4;
    int row = nt * 16 + n;
    const float* src = (g ? W_hh : W_ih) + ((size_t)l * G3 + row) * HID;
    unsigned* dst = wsw + (size_t)e * 8;
#pragma unroll
    for (int v = 0; v < 8; ++v) {
      int k = kb * 32 + ((v >> 2) << 4) + hi * 8 + ((v & 3) << 1);
      dst[v] = pack2bf(src[k], src[k + 1]);
    }
  }

  // Attention weights Wa[l][h][a]: B[k=h][n=a]; 4 x 16 ntile x 16 kb x 32 lanes
  const int NAWE = NLAYER * 16 * 16 * 32;              // 32,768 entries
  for (int e = gid; e < NAWE; e += stride) {
    int lane = e & 31;
    int kb = (e >> 5) & 15;
    int nt = (e >> 9) & 15;
    int l = e >> 13;
    int n = lane & 15, hi = lane >> 4;
    int a = nt * 16 + n;
    const float* src = Wa + (size_t)l * HID * ADIM;
    unsigned* dst = wasw + (size_t)e * 8;
#pragma unroll
    for (int v = 0; v < 8; ++v) {
      int k = kb * 32 + ((v >> 2) << 4) + hi * 8 + ((v & 3) << 1);
      dst[v] = pack2bf(src[(size_t)k * ADIM + a], src[(size_t)(k + 1) * ADIM + a]);
    }
  }

  for (int i = gid; i < NLAYER * BH; i += stride) { hf[i] = 0.0f; hbf[i] = 0; }
}

// ---------------- persistent recurrence kernel ------------------------------------
__global__ void __launch_bounds__(BLOCK) fbrnn_persistent(
    const int* __restrict__ tokens, const float* __restrict__ emb,
    const float* __restrict__ b_ih, const float* __restrict__ b_hh,
    const float* __restrict__ ba, const float* __restrict__ va,
    const unsigned* __restrict__ wsw, const unsigned* __restrict__ wasw,
    float* __restrict__ hf, unsigned short* __restrict__ hbf,
    float* __restrict__ newhf, unsigned short* __restrict__ newhbf,
    float* __restrict__ proj, float* __restrict__ aw,
    unsigned* __restrict__ cnt,
    float* __restrict__ out) {
  __shared__ unsigned short ldsA[2][16 * LPITCH];  // [0]=layer input, [1]=hidden

  const int tid = threadIdx.x;
  const int bid = blockIdx.x;
  const int gtid = bid * BLOCK + tid;
  const int wvin = tid >> 5;          // wave in block: 0..7
  const int wv = gtid >> 5;           // global wave:  0..127
  const int lane = tid & 31;
  const int mt = bid >> 2;            // batch tile 0..3 (16 rows each)
  const int nc = ((bid & 3) << 3) + wvin;  // column group 0..31 (16 cols each)
  unsigned bgen = 0;

  for (int t = 0; t < TSTEPS; ++t) {
    // ============ 4 GRU layers: fused GEMM(6 tiles) + gates, 1 barrier each =====
    for (int l = 0; l < NLAYER; ++l) {
      // --- stage A tiles (16x512 bf16 each) into LDS ---
      if (l == 0) {
        // embedding gather + f32->bf16 (needs conversion -> VALU path)
        for (int idx = tid; idx < 16 * EMB; idx += BLOCK) {
          int m = idx >> 9, k = idx & 511;
          int tok = tokens[t * BATCH + mt * 16 + m];
          ldsA[0][m * LPITCH + k] = f2bf(emb[(size_t)tok * EMB + k]);
        }
      } else {
        const unsigned short* src = newhbf + (size_t)(l - 1) * BH + mt * 16 * HID;
#ifdef USE_ASYNC_LDS
        for (int c = tid; c < 16 * 64; c += BLOCK) {       // 16B chunks
          int m = c >> 6, kc = c & 63;
          __builtin_amdgcn_global_load_async_to_lds_b128(
              to_g128(src + m * HID + kc * 8),
              to_l128(&ldsA[0][m * LPITCH + kc * 8]), 0, 0);
        }
#else
        for (int idx = tid; idx < 16 * (HID / 2); idx += BLOCK) {
          int m = idx >> 8, k2 = idx & 255;
          *reinterpret_cast<unsigned*>(&ldsA[0][m * LPITCH + k2 * 2]) =
              ((const unsigned*)src)[m * (HID / 2) + k2];
        }
#endif
      }
      {
        const unsigned short* src = hbf + (size_t)l * BH + mt * 16 * HID;
#ifdef USE_ASYNC_LDS
        for (int c = tid; c < 16 * 64; c += BLOCK) {       // 16B chunks
          int m = c >> 6, kc = c & 63;
          __builtin_amdgcn_global_load_async_to_lds_b128(
              to_g128(src + m * HID + kc * 8),
              to_l128(&ldsA[1][m * LPITCH + kc * 8]), 0, 0);
        }
#else
        for (int idx = tid; idx < 16 * (HID / 2); idx += BLOCK) {
          int m = idx >> 8, k2 = idx & 255;
          *reinterpret_cast<unsigned*>(&ldsA[1][m * LPITCH + k2 * 2]) =
              ((const unsigned*)src)[m * (HID / 2) + k2];
        }
#endif
      }
#ifdef USE_ASYNC_LDS
      wait_async0();
#endif
      __syncthreads();

      // --- 6 accumulators: (i,h) x (r,z,n) gates for 16 rows x 16 cols ---
      v8f air = {0.f,0.f,0.f,0.f,0.f,0.f,0.f,0.f};
      v8f aiz = air, ain = air, ahr = air, ahz = air, ahn = air;
      const int base_i = ((l * 2 + 0) * 96) * 16;
      const int base_h = ((l * 2 + 1) * 96) * 16;
      for (int kb = 0; kb < HID / 32; ++kb) {
        v16bf aI = load_fragA_lds(&ldsA[0][0], kb * 32, lane);
        v16bf aH = load_fragA_lds(&ldsA[1][0], kb * 32, lane);
        v16bf bir = load_fragW(wsw, base_i + (nc     ) * 16 + kb, lane);
        v16bf biz = load_fragW(wsw, base_i + (nc + 32) * 16 + kb, lane);
        v16bf bin = load_fragW(wsw, base_i + (nc + 64) * 16 + kb, lane);
        v16bf bhr = load_fragW(wsw, base_h + (nc     ) * 16 + kb, lane);
        v16bf bhz = load_fragW(wsw, base_h + (nc + 32) * 16 + kb, lane);
        v16bf bhn = load_fragW(wsw, base_h + (nc + 64) * 16 + kb, lane);
        air = wmma_bf16(aI, bir, air);
        aiz = wmma_bf16(aI, biz, aiz);
        ain = wmma_bf16(aI, bin, ain);
        ahr = wmma_bf16(aH, bhr, ahr);
        ahz = wmma_bf16(aH, bhz, ahz);
        ahn = wmma_bf16(aH, bhn, ahn);
      }

      // --- in-register GRU gates (torch order r,z,n) ---
      {
        const int n_ = lane & 15, hi = lane >> 4;
        const int j = nc * 16 + n_;
        const float bir_ = b_ih[l * G3 + j];
        const float biz_ = b_ih[l * G3 + HID + j];
        const float bin_ = b_ih[l * G3 + 2 * HID + j];
        const float bhr_ = b_hh[l * G3 + j];
        const float bhz_ = b_hh[l * G3 + HID + j];
        const float bhn_ = b_hh[l * G3 + 2 * HID + j];
#pragma unroll
        for (int r = 0; r < 8; ++r) {
          const int m = mt * 16 + hi * 8 + r;
          const int hidx = l * BH + m * HID + j;
          float rr = sigm(air[r] + bir_ + ahr[r] + bhr_);
          float zz = sigm(aiz[r] + biz_ + ahz[r] + bhz_);
          float nn = tanhf(ain[r] + bin_ + rr * (ahn[r] + bhn_));
          float hp = hf[hidx];
          float nv = (1.0f - zz) * nn + zz * hp;
          newhf[hidx] = nv;
          newhbf[hidx] = f2bf(nv);
        }
      }
      grid_sync(cnt, &bgen);
    }

    // ============ attention projections: 10 (i,k>=i) GEMMs [64x256,K=512] ========
    for (int tile = wv; tile < 640; tile += NWAVES) {
      const int c = tile >> 6, r2 = tile & 63;
      const int pmt = r2 >> 4, nt = r2 & 15;
      int ii, kk;
      if (c < 4)      { ii = 0; kk = c; }
      else if (c < 7) { ii = 1; kk = c - 3; }
      else if (c < 9) { ii = 2; kk = c - 5; }
      else            { ii = 3; kk = 3; }
      const unsigned short* At = newhbf + (size_t)kk * BH + pmt * 16 * HID;
      v8f acc = {0.f,0.f,0.f,0.f,0.f,0.f,0.f,0.f};
      for (int kb = 0; kb < HID / 32; ++kb) {
        v16bf a = load_fragA_g(At, HID, kb * 32, lane);
        v16bf b = load_fragW(wasw, (ii * 16 + nt) * 16 + kb, lane);
        acc = wmma_bf16(a, b, acc);
      }
      const int n_ = lane & 15, hi = lane >> 4;
      const int ncol = nt * 16 + n_;
      const float bv = ba[ii * ADIM + ncol];
      float* o = proj + (ii * 4 + kk) * (BATCH * ADIM);
#pragma unroll
      for (int r = 0; r < 8; ++r) {
        int m = pmt * 16 + hi * 8 + r;
        o[m * ADIM + ncol] = tanhf(acc[r] + bv);
      }
    }
    grid_sync(cnt, &bgen);

    // ============ e = proj . va ; softmax over k (one thread per (i,b)) ==========
    if (gtid < NLAYER * BATCH) {
      int i = gtid >> 6, b = gtid & 63;
      float ev[NLAYER];
      float emax = -1e30f;
#pragma unroll
      for (int k = 0; k < NLAYER; ++k) {
        if (k >= i) {
          const float* pr = proj + (i * 4 + k) * (BATCH * ADIM) + b * ADIM;
          const float* vv = va + i * ADIM;
          float s = 0.f;
          for (int a2 = 0; a2 < ADIM; ++a2) s += pr[a2] * vv[a2];
          ev[k] = s;
          emax = fmaxf(emax, s);
        } else ev[k] = 0.f;
      }
      float den = 0.f;
#pragma unroll
      for (int k = 0; k < NLAYER; ++k)
        if (k >= i) { ev[k] = __expf(ev[k] - emax); den += ev[k]; }
      float rden = 1.f / den;
#pragma unroll
      for (int k = 0; k < NLAYER; ++k)
        if (k >= i) aw[(i * 4 + k) * BATCH + b] = ev[k] * rden;
    }
    grid_sync(cnt, &bgen);

    // ============ weighted mix -> next hidden (i==3 is the step output) ==========
    for (int idx = gtid; idx < NLAYER * BH; idx += NTHREADS) {
      int i = idx >> 15, rem = idx & (BH - 1);
      int b = rem >> 9;
      float s = 0.f;
#pragma unroll
      for (int k = 0; k < NLAYER; ++k)
        if (k >= i) s += aw[(i * 4 + k) * BATCH + b] * newhf[k * BH + rem];
      hf[i * BH + rem] = s;
      hbf[i * BH + rem] = f2bf(s);
      if (i == 3) out[(size_t)t * BH + rem] = s;
    }
    grid_sync(cnt, &bgen);
  }
}

extern "C" void kernel_launch(void* const* d_in, const int* in_sizes, int n_in,
                              void* d_out, int out_size, void* d_ws, size_t ws_size,
                              hipStream_t stream) {
  (void)in_sizes; (void)n_in; (void)out_size; (void)ws_size;
  const int*   tokens = (const int*)  d_in[0];
  const float* emb    = (const float*)d_in[1];
  const float* W_ih   = (const float*)d_in[2];
  const float* W_hh   = (const float*)d_in[3];
  const float* b_ih   = (const float*)d_in[4];
  const float* b_hh   = (const float*)d_in[5];
  const float* Wa     = (const float*)d_in[6];
  const float* ba     = (const float*)d_in[7];
  const float* va     = (const float*)d_in[8];
  float* out = (float*)d_out;

  char* ws = (char*)d_ws;
  size_t off = 0;
  auto take = [&](size_t bytes) -> void* {
    off = (off + 255) & ~(size_t)255;
    void* p = ws + off;
    off += bytes;
    return p;
  };

  unsigned*       cnt    = (unsigned*)      take(256);
  unsigned*       wsw    = (unsigned*)      take((size_t)NLAYER * 2 * 96 * 16 * 32 * 8 * 4); // 12.6MB
  unsigned*       wasw   = (unsigned*)      take((size_t)NLAYER * 16 * 16 * 32 * 8 * 4);     // 1MB
  float*          hf     = (float*)         take((size_t)NLAYER * BH * 4);
  unsigned short* hbf    = (unsigned short*)take((size_t)NLAYER * BH * 2);
  float*          newhf  = (float*)         take((size_t)NLAYER * BH * 4);
  unsigned short* newhbf = (unsigned short*)take((size_t)NLAYER * BH * 2);
  float*          proj   = (float*)         take((size_t)16 * BATCH * ADIM * 4);
  float*          aw     = (float*)         take((size_t)16 * BATCH * 4);

  fbrnn_prep<<<1024, 256, 0, stream>>>(W_ih, W_hh, Wa, wsw, wasw, hf, hbf, cnt);
  fbrnn_persistent<<<NBLK, BLOCK, 0, stream>>>(
      tokens, emb, b_ih, b_hh, ba, va, wsw, wasw,
      hf, hbf, newhf, newhbf, proj, aw, cnt, out);
}
